// BilinearAttention_49134425866840
// MI455X (gfx1250) — matrix operand order repaired and checked by reading, not compiled, not run on hardware
//
#include <hip/hip_runtime.h>
#include <hip/hip_bf16.h>
#include <math.h>

// Problem constants
#define N_NODES 3072
#define DIN     1024
#define DEGO    64
#define DLOC    16
#define DGLB    16
#define DOUTN   1024
#define DSC     112   // ego(64) + q_local(16) + k_local(16) + q_global(16)
#define DAT     96    // ego(64) + local(16) + global(16)

typedef __attribute__((ext_vector_type(16))) __bf16 v16bf;
typedef __attribute__((ext_vector_type(8)))  float  v8f;
typedef __attribute__((ext_vector_type(4)))  unsigned int u32x4;
typedef __attribute__((ext_vector_type(8)))  int i32x8;
typedef __attribute__((ext_vector_type(4)))  int i32x4;

#if defined(__has_builtin)
#if __has_builtin(__builtin_amdgcn_tensor_load_to_lds) && \
    __has_builtin(__builtin_amdgcn_s_wait_tensorcnt)
#define HAVE_TDM 1
#endif
#endif
#if __has_include(<hip/amd_detail/amd_gfx1250_TDM.h>)
#define TDM_6ARG 1   // therock-10.0 headers -> 6-arg builtin
#endif

__device__ __forceinline__ float nonneg(float x) {
  // elu(x)+1 : x>=0 -> x+1 ; x<0 -> exp(x)
  return x >= 0.f ? x + 1.f : __expf(x);
}

// split f32 into bf16 hi + bf16 residual (2-term split-bf16)
#define BSPLIT(F, VH, VL, I) { float _f = (F); __bf16 _h = (__bf16)_f; \
                               VH[I] = _h; VL[I] = (__bf16)(_f - (float)_h); }

// ---------------------------------------------------------------------------
// x_bar reduction, two-phase (deterministic, no atomics)
// ---------------------------------------------------------------------------
__global__ __launch_bounds__(256)
void xbar_partial_kernel(const float* __restrict__ x, float* __restrict__ part) {
  int b = blockIdx.x;                                   // 96 blocks, 32 rows each
  for (int c = threadIdx.x; c < DIN; c += 256) {
    float s = 0.f;
    const float* base = x + (size_t)b * 32 * DIN + c;
    #pragma unroll 4
    for (int r = 0; r < 32; ++r) s += base[(size_t)r * DIN];
    part[b * DIN + c] = s;
  }
}

__global__ __launch_bounds__(256)
void xbar_final_kernel(const float* __restrict__ part, float* __restrict__ xacc) {
  int c = blockIdx.x * 256 + threadIdx.x;               // 4 blocks
  float s = 0.f;
  for (int b = 0; b < 96; ++b) s += part[b * DIN + c];
  xacc[c] = s;                                          // sum; /N folded later
}

// ---------------------------------------------------------------------------
// Weight preprocessing: concatenated score matrix Wcat [112 x 1024] ([j][k])
// and value matrix Vcat [1024 x 96] ([o][c]) as split-bf16. 1/d folded into
// Wcat so local/global q*k products carry 1/d^2 and ego scores carry 1/d.
// ---------------------------------------------------------------------------
__global__ __launch_bounds__(256)
void prep_w_kernel(const float* __restrict__ w_ego, const float* __restrict__ shiftp,
                   const float* __restrict__ w_ql,  const float* __restrict__ w_kl,
                   const float* __restrict__ w_vl,  const float* __restrict__ w_qg,
                   const float* __restrict__ w_vg,
                   __bf16* __restrict__ Wh, __bf16* __restrict__ Wl,
                   __bf16* __restrict__ Vh, __bf16* __restrict__ Vl) {
  const float inv_d = 1.0f / (float)DIN;
  int idx = blockIdx.x * 256 + threadIdx.x;             // 448*256 = 114688 = 112*1024
  {
    int j = idx / DIN, k = idx % DIN;
    float v;
    if (j < 64)       v = (nonneg(w_ego[k * 64 + j]) + shiftp[0]) * inv_d;  // ego
    else if (j < 80)  v = w_ql[(j - 64) * DIN + k] * inv_d;                 // q_local
    else if (j < 96)  v = nonneg(w_kl[(j - 80) * DIN + k]) * inv_d;         // k_local
    else              v = w_qg[(j - 96) * DIN + k] * inv_d;                 // q_global
    __bf16 h = (__bf16)v;
    Wh[idx] = h; Wl[idx] = (__bf16)(v - (float)h);
  }
  if (idx < DOUTN * DAT) {
    int o = idx / DAT, c = idx % DAT;
    float v;
    if (c < 64)      v = nonneg(w_ego[o * 64 + c]);          // v_ego_w^T (no shift)
    else if (c < 80) v = nonneg(w_vl[o * 16 + (c - 64)]);
    else             v = nonneg(w_vg[o * 16 + (c - 80)]);
    __bf16 h = (__bf16)v;
    Vh[idx] = h; Vl[idx] = (__bf16)(v - (float)h);
  }
}

// k_global scaled: kg[c] = (1/d) * sum_k (xacc[k]/N) * nonneg(w_kglobal[c][k])
__global__ __launch_bounds__(256)
void kglobal_kernel(const float* __restrict__ xacc, const float* __restrict__ wkg,
                    float* __restrict__ kg) {
  __shared__ float red[256];
  int t = threadIdx.x, c = t >> 4, l = t & 15;
  float s = 0.f;
  for (int k = l; k < DIN; k += 16)
    s += (xacc[k] * (1.0f / (float)N_NODES)) * nonneg(wkg[c * DIN + k]);
  red[t] = s;
  __syncthreads();
  if (l == 0) {
    float tot = 0.f;
    #pragma unroll
    for (int i = 0; i < 16; ++i) tot += red[(c << 4) + i];
    kg[c] = tot * (1.0f / (float)DIN);
  }
}

// ---------------------------------------------------------------------------
// Fused score GEMM: S[3072,112] = x[3072,1024] @ Wcat[1024,112]
// split-bf16 (hi*hi + lo*hi + hi*lo) on v_wmma_f32_16x16x32_bf16.
// ---------------------------------------------------------------------------
__global__ __launch_bounds__(256)
void score_gemm_kernel(const float* __restrict__ x,
                       const __bf16* __restrict__ wh, const __bf16* __restrict__ wl,
                       float* __restrict__ S) {
  int wave = threadIdx.x >> 5, lane = threadIdx.x & 31;
  int tile = blockIdx.x * 8 + wave;
  int mt = tile / 7, nt = tile % 7;
  int grp = lane >> 4, l16 = lane & 15;
  const float*  xr = x  + (size_t)(mt * 16 + l16) * DIN;   // A: lane holds row M
  const __bf16* bh = wh + (size_t)(nt * 16 + l16) * DIN;   // B: lane holds col N
  const __bf16* bl = wl + (size_t)(nt * 16 + l16) * DIN;
  v8f acc = {};
  for (int k0 = 0; k0 < DIN; k0 += 32) {
    int ka = k0 + grp * 16;                // A 16-bit layout: half-wave K-split
    const float4 a0 = *(const float4*)(xr + ka);
    const float4 a1 = *(const float4*)(xr + ka + 4);
    const float4 a2 = *(const float4*)(xr + ka + 8);
    const float4 a3 = *(const float4*)(xr + ka + 12);
    v16bf ah, al;
    BSPLIT(a0.x, ah, al, 0)  BSPLIT(a0.y, ah, al, 1)
    BSPLIT(a0.z, ah, al, 2)  BSPLIT(a0.w, ah, al, 3)
    BSPLIT(a1.x, ah, al, 4)  BSPLIT(a1.y, ah, al, 5)
    BSPLIT(a1.z, ah, al, 6)  BSPLIT(a1.w, ah, al, 7)
    BSPLIT(a2.x, ah, al, 8)  BSPLIT(a2.y, ah, al, 9)
    BSPLIT(a2.z, ah, al, 10) BSPLIT(a2.w, ah, al, 11)
    BSPLIT(a3.x, ah, al, 12) BSPLIT(a3.y, ah, al, 13)
    BSPLIT(a3.z, ah, al, 14) BSPLIT(a3.w, ah, al, 15)
    int kb = k0 + grp * 16;                // B: 16 consecutive K per half-wave
    v16bf bhv = *(const v16bf*)(bh + kb);
    v16bf blv = *(const v16bf*)(bl + kb);
    acc = __builtin_amdgcn_wmma_f32_16x16x32_bf16(false, ah, false, bhv, (short)0, acc, false, false);
    acc = __builtin_amdgcn_wmma_f32_16x16x32_bf16(false, al, false, bhv, (short)0, acc, false, false);
    acc = __builtin_amdgcn_wmma_f32_16x16x32_bf16(false, ah, false, blv, (short)0, acc, false, false);
  }
  int rbase = mt * 16 + grp * 8, col = nt * 16 + l16;
  #pragma unroll
  for (int i = 0; i < 8; ++i)
    S[(size_t)(rbase + i) * DSC + col] = acc[i];
}

// ---------------------------------------------------------------------------
// Local attention + softmax fusion. One wave per row n.
// Entire k_local table (3072x16 f32 = 192KB) lives in LDS, filled by the
// Tensor Data Mover (one tensor_load_to_lds, TENSORcnt-tracked) when
// available; cooperative loop fallback otherwise.
// ---------------------------------------------------------------------------
__global__ __launch_bounds__(256)
void local_attn_kernel(const float* __restrict__ S,
                       const unsigned char* __restrict__ adj,
                       const float* __restrict__ kg,
                       float* __restrict__ attn) {
  extern __shared__ float lk[];                         // [3072*16] at LDS offset 0
#if defined(HAVE_TDM)
  if (threadIdx.x < 32) {                               // one wave issues the DMA
    unsigned long long ga = (unsigned long long)(const void*)(S + 80); // tile start
    u32x4 g0; i32x8 g1;
    i32x4 g2 = {0, 0, 0, 0}, g3 = {0, 0, 0, 0};         // 2D tensor: groups 2/3 null
    g0[0] = 1u;                                         // count=1, user descriptor
    g0[1] = 0u;                                         // lds_addr = 0 (lk[] base)
    g0[2] = (unsigned)(ga & 0xffffffffu);               // global_addr[31:0]
    g0[3] = (unsigned)((ga >> 32) & 0x1ffffffu) | 0x80000000u; // addr[56:32] | type=2
    g1[0] = 0x00020000;        // workgroup_mask=0, data_size=2 (4 bytes)
    g1[1] = 0;                 // abar=0, tensor_dim0[15:0]=0 (dim0 = 1<<20)
    g1[2] = 0x00000010;        // tensor_dim0[31:16]=0x10, tensor_dim1[15:0]=0
    g1[3] = 0x00100010;        // tensor_dim1[31:16]=0x10, tile_dim0=16
    g1[4] = N_NODES;           // tile_dim1=3072, tile_dim2=0
    g1[5] = DSC;               // tensor_dim0_stride = 112 (elements)
    g1[6] = 0; g1[7] = 0;
#if defined(TDM_6ARG)
    i32x8 g4 = {0, 0, 0, 0, 0, 0, 0, 0};
    __builtin_amdgcn_tensor_load_to_lds(g0, g1, g2, g3, g4, 0);
#else
    __builtin_amdgcn_tensor_load_to_lds(g0, g1, g2, g3, 0);
#endif
    __builtin_amdgcn_s_wait_tensorcnt(0);
  }
  __syncthreads();
#else
  for (int idx = threadIdx.x; idx < N_NODES * DLOC; idx += 256) {
    int m = idx >> 4, c = idx & 15;
    lk[idx] = S[(size_t)m * DSC + 80 + c];              // k_local (pre-scaled by 1/d)
  }
  __syncthreads();
#endif
  int lane = threadIdx.x & 31, wave = threadIdx.x >> 5;
  int c = lane & 15, grp = lane >> 4;
  float kgv = kg[c];
  for (int it = 0; it < 4; ++it) {
    int n = blockIdx.x * 32 + it * 8 + wave;
    const float* srow = S + (size_t)n * DSC;
    const unsigned char* arow = adj + (size_t)n * N_NODES;
    __builtin_prefetch(arow, 0, 0);                     // global_prefetch_b8
    float q = srow[64 + c];
    // ---- pass 1: masked max of q*k over adjacent m ----
    float mx = -1e9f;                                   // matches reference mask fill
    #pragma unroll 8
    for (int m = grp; m < N_NODES; m += 2) {
      float qk = q * lk[m * 16 + c];
      if (arow[m]) mx = fmaxf(mx, qk);
    }
    float e0 = srow[lane], e1 = srow[32 + lane];        // 64 ego scores (2/lane)
    float g  = srow[96 + c] * kgv;                      // global score (dup per half)
    float rmax = fmaxf(fmaxf(e0, e1), fmaxf(mx, g));
    #pragma unroll
    for (int off = 16; off >= 1; off >>= 1)
      rmax = fmaxf(rmax, __shfl_xor(rmax, off, 32));    // row max_score
    // ---- pass 2: masked exp sum (one v_exp per element) ----
    __builtin_prefetch(arow, 0, 0);
    float ss = 0.f;
    #pragma unroll 8
    for (int m = grp; m < N_NODES; m += 2) {
      float qk = q * lk[m * 16 + c];
      if (arow[m]) ss += __expf(qk - rmax);
    }
    ss += __shfl_xor(ss, 16, 32);                       // per-c e_local (dup)
    float eg0 = __expf(e0 - rmax), eg1 = __expf(e1 - rmax);
    float egl = __expf(g - rmax);
    float part = eg0 + eg1 + (grp == 0 ? (ss + egl) : 0.f);
    #pragma unroll
    for (int off = 16; off >= 1; off >>= 1)
      part += __shfl_xor(part, off, 32);                // sum_score
    float inv = 1.0f / part;
    float* ar = attn + (size_t)n * DAT;
    ar[lane]      = eg0 * inv;                          // ego_attn 0..63
    ar[32 + lane] = eg1 * inv;
    if (grp == 0) {
      ar[64 + c] = ss  * inv;                           // local_attn
      ar[80 + c] = egl * inv;                           // global_attn
    }
  }
}

// ---------------------------------------------------------------------------
// Fused output GEMM: out[3072,1024] = attn[3072,96] @ Vcat[96,1024] + nonneg(bias)
// ---------------------------------------------------------------------------
__global__ __launch_bounds__(256)
void out_gemm_kernel(const float* __restrict__ attn,
                     const __bf16* __restrict__ vh, const __bf16* __restrict__ vl,
                     const float* __restrict__ bias, float* __restrict__ out) {
  int wave = threadIdx.x >> 5, lane = threadIdx.x & 31;
  int tile = blockIdx.x * 8 + wave;
  int mt = tile >> 6, nt = tile & 63;
  int grp = lane >> 4, l16 = lane & 15;
  const float* ar = attn + (size_t)(mt * 16 + l16) * DAT;
  int o = nt * 16 + l16;
  const __bf16* bh = vh + (size_t)o * DAT;
  const __bf16* bl = vl + (size_t)o * DAT;
  v8f acc = {};
  #pragma unroll
  for (int k0 = 0; k0 < DAT; k0 += 32) {
    int ka = k0 + grp * 16;
    const float4 a0 = *(const float4*)(ar + ka);
    const float4 a1 = *(const float4*)(ar + ka + 4);
    const float4 a2 = *(const float4*)(ar + ka + 8);
    const float4 a3 = *(const float4*)(ar + ka + 12);
    v16bf ah, al;
    BSPLIT(a0.x, ah, al, 0)  BSPLIT(a0.y, ah, al, 1)
    BSPLIT(a0.z, ah, al, 2)  BSPLIT(a0.w, ah, al, 3)
    BSPLIT(a1.x, ah, al, 4)  BSPLIT(a1.y, ah, al, 5)
    BSPLIT(a1.z, ah, al, 6)  BSPLIT(a1.w, ah, al, 7)
    BSPLIT(a2.x, ah, al, 8)  BSPLIT(a2.y, ah, al, 9)
    BSPLIT(a2.z, ah, al, 10) BSPLIT(a2.w, ah, al, 11)
    BSPLIT(a3.x, ah, al, 12) BSPLIT(a3.y, ah, al, 13)
    BSPLIT(a3.z, ah, al, 14) BSPLIT(a3.w, ah, al, 15)
    int kb = k0 + grp * 16;
    v16bf bhv = *(const v16bf*)(bh + kb);
    v16bf blv = *(const v16bf*)(bl + kb);
    acc = __builtin_amdgcn_wmma_f32_16x16x32_bf16(false, ah, false, bhv, (short)0, acc, false, false);
    acc = __builtin_amdgcn_wmma_f32_16x16x32_bf16(false, al, false, bhv, (short)0, acc, false, false);
    acc = __builtin_amdgcn_wmma_f32_16x16x32_bf16(false, ah, false, blv, (short)0, acc, false, false);
  }
  float bnn = nonneg(bias[o]);
  int rbase = mt * 16 + grp * 8;
  #pragma unroll
  for (int i = 0; i < 8; ++i)
    out[(size_t)(rbase + i) * DOUTN + o] = acc[i] + bnn;
}

// ---------------------------------------------------------------------------
extern "C" void kernel_launch(void* const* d_in, const int* in_sizes, int n_in,
                              void* d_out, int out_size, void* d_ws, size_t ws_size,
                              hipStream_t stream) {
  (void)in_sizes; (void)n_in; (void)out_size; (void)ws_size;
  const unsigned char* adj = (const unsigned char*)d_in[0];  // jax bool = 1 byte
  const float* x     = (const float*)d_in[1];
  const float* w_ego = (const float*)d_in[2];
  const float* shift = (const float*)d_in[3];
  const float* w_ql  = (const float*)d_in[4];
  const float* w_kl  = (const float*)d_in[5];
  const float* w_vl  = (const float*)d_in[6];
  const float* w_qg  = (const float*)d_in[7];
  const float* w_kg  = (const float*)d_in[8];
  const float* w_vg  = (const float*)d_in[9];
  const float* bias  = (const float*)d_in[10];
  float* out = (float*)d_out;

  char* ws = (char*)d_ws;                    // total footprint ~3.9 MB
  float*  S    = (float*)(ws + 0x000000);    // [3072,112]
  float*  attn = (float*)(ws + 0x160000);    // [3072,96]
  float*  xprt = (float*)(ws + 0x280000);    // [96,1024]
  float*  xacc = (float*)(ws + 0x300000);    // [1024]
  float*  kg   = (float*)(ws + 0x301000);    // [16]
  __bf16* Wh   = (__bf16*)(ws + 0x302000);   // [112,1024]
  __bf16* Wl   = (__bf16*)(ws + 0x340000);
  __bf16* Vh   = (__bf16*)(ws + 0x380000);   // [1024,96]
  __bf16* Vl   = (__bf16*)(ws + 0x3B0000);

  xbar_partial_kernel<<<96, 256, 0, stream>>>(x, xprt);
  xbar_final_kernel<<<4, 256, 0, stream>>>(xprt, xacc);
  prep_w_kernel<<<448, 256, 0, stream>>>(w_ego, shift, w_ql, w_kl, w_vl, w_qg, w_vg,
                                         Wh, Wl, Vh, Vl);
  kglobal_kernel<<<1, 256, 0, stream>>>(xacc, w_kg, kg);
  score_gemm_kernel<<<168, 256, 0, stream>>>(x, Wh, Wl, S);
  local_attn_kernel<<<96, 256, N_NODES * DLOC * sizeof(float), stream>>>(S, adj, kg, attn);
  out_gemm_kernel<<<1536, 256, 0, stream>>>(attn, Vh, Vl, bias, out);
}
